// S5Regressor_83580063580537
// MI455X (gfx1250) — matrix operand order repaired and checked by reading, not compiled, not run on hardware
//
#include <hip/hip_runtime.h>
#include <hip/hip_bf16.h>
#include <math.h>

#define BDIM 8
#define LSEQ 4096
#define DIN 64
#define DMODEL 512
#define PDIM 512
#define NLAYER 3
#define MROWS (BDIM * LSEQ) /* 32768 */
#define CHUNK 64
#define NCH (LSEQ / CHUNK) /* 64 */

typedef __attribute__((ext_vector_type(16))) __bf16 bf16x16;
typedef __attribute__((ext_vector_type(8))) float f32x8;
typedef __attribute__((ext_vector_type(4))) unsigned int u32x4;
typedef int gvec4i __attribute__((vector_size(16))); // matches builtin param

union FragBF { u32x4 q[2]; bf16x16 v; };

// CDNA5 async global->LDS path (ASYNCcnt), guarded so the file always compiles
#if defined(__HIP_DEVICE_COMPILE__) &&                                         \
    __has_builtin(__builtin_amdgcn_global_load_async_to_lds_b128) &&           \
    __has_builtin(__builtin_amdgcn_s_wait_asynccnt)
#define USE_ASYNC 1
#else
#define USE_ASYNC 0
#endif

#define AS_GLOBAL __attribute__((address_space(1)))
#define AS_LOCAL __attribute__((address_space(3)))

__device__ __forceinline__ unsigned short f2bf(float f) {
  unsigned u = __float_as_uint(f);
  u = (u + 0x7fffu + ((u >> 16) & 1u)) >> 16;
  return (unsigned short)u;
}
__device__ __forceinline__ float gelu_exact(float x) {
  return 0.5f * x * (1.0f + erff(x * 0.70710678118654752f));
}

// ---------------------------------------------------------------------------
// bf16 WMMA GEMM: C(M,N) = A(M,K) @ W(N,K)^T ; A and W both bf16 row-major.
// Block tile 128(M) x 64(N), K-step 32, 256 threads = 8 waves, each wave a
// 32x32 patch as 2x2 v_wmma_f32_16x16x32_bf16 tiles. Double-buffered LDS with
// async global->LDS staging (fallback: register copy).
// epi: 0 = C = acc ; 1 = C = acc + bias[n] ; 3 = hio += acc + add2
// ---------------------------------------------------------------------------
__global__ __launch_bounds__(256) void gemm_bf16_wmma(
    const unsigned short* __restrict__ Ab, const unsigned short* __restrict__ W,
    float* __restrict__ C, int M, int N, int K, int epi,
    const float* __restrict__ bias, const float* __restrict__ add2,
    float* __restrict__ hio) {
  __shared__ __align__(16) unsigned short lA[2][128 * 40]; // padded rows (80B)
  __shared__ __align__(16) unsigned short lB[2][64 * 40];

  const int tid = threadIdx.x;
  const int nb = N >> 6;
  const int tm = (int)blockIdx.x / nb;
  const int tn = (int)blockIdx.x % nb;
  const int m_base = tm << 7;
  const int n_base = tn << 6;
  const int lane = tid & 31, wid = tid >> 5;
  const int wm = wid & 3;  // 4 M-subtiles of 32 rows
  const int wn = wid >> 2; // 2 N-subtiles of 32 cols
  const int lm = lane & 15, hi = lane >> 4;
  const int KT = K >> 5;

  // staging coords: A = 128 rows x 2 segs of 16 bf16 (256 xfers, 1/thread)
  //                 B = 64 rows x 2 segs (128 xfers, threads < 128)
  const int s_arow = tid >> 1, s_aseg = tid & 1;
  const int s_brow = (tid & 127) >> 1, s_bseg = tid & 1;
  const bool doB = tid < 128;

  f32x8 acc[2][2];
#pragma unroll
  for (int i = 0; i < 2; i++)
#pragma unroll
    for (int j = 0; j < 2; j++)
#pragma unroll
      for (int r = 0; r < 8; r++) acc[i][j][r] = 0.0f;

  auto stageTile = [&](int kt, int buf) {
    const int k0 = kt << 5;
    const unsigned short* ga =
        Ab + (size_t)(m_base + s_arow) * K + k0 + s_aseg * 16;
    unsigned short* la = &lA[buf][s_arow * 40 + s_aseg * 16];
#if USE_ASYNC
    __builtin_amdgcn_global_load_async_to_lds_b128(
        (AS_GLOBAL gvec4i*)const_cast<unsigned short*>(ga),
        (AS_LOCAL gvec4i*)la, 0, 0);
#else
    *(u32x4*)la = *(const u32x4*)ga;
#endif
    if (doB) {
      const unsigned short* gb =
          W + (size_t)(n_base + s_brow) * K + k0 + s_bseg * 16;
      unsigned short* lb = &lB[buf][s_brow * 40 + s_bseg * 16];
#if USE_ASYNC
      __builtin_amdgcn_global_load_async_to_lds_b128(
          (AS_GLOBAL gvec4i*)const_cast<unsigned short*>(gb),
          (AS_LOCAL gvec4i*)lb, 0, 0);
#else
      *(u32x4*)lb = *(const u32x4*)gb;
#endif
    }
  };

  stageTile(0, 0);

  for (int kt = 0; kt < KT; ++kt) {
    const int cur = kt & 1;
#if USE_ASYNC
    __builtin_amdgcn_s_wait_asynccnt(0);
#endif
    __syncthreads(); // tile `cur` resident; all reads of `cur^1` finished
    if (kt + 1 < KT) stageTile(kt + 1, cur ^ 1);

    // ---- fragments (contiguous 16B chunks matching ISA lane layouts)
    FragBF a0, a1, b0, b1;
    {
      // A 16x32: lanes 0-15 -> K {0..7,16..23}; lanes 16-31 -> {8..15,24..31}
      const unsigned short* pa0 = &lA[cur][(wm * 32 + 0 * 16 + lm) * 40 + hi * 8];
      a0.q[0] = *(const u32x4*)pa0;
      a0.q[1] = *(const u32x4*)(pa0 + 16);
      const unsigned short* pa1 = &lA[cur][(wm * 32 + 1 * 16 + lm) * 40 + hi * 8];
      a1.q[0] = *(const u32x4*)pa1;
      a1.q[1] = *(const u32x4*)(pa1 + 16);
      // B 32x16: lane n holds col n; lanes 0-15 K {0..7,16..23}, lanes 16-31 K {8..15,24..31}
      const unsigned short* pb0 = &lB[cur][(wn * 32 + 0 * 16 + lm) * 40 + hi * 8];
      b0.q[0] = *(const u32x4*)pb0;
      b0.q[1] = *(const u32x4*)(pb0 + 16);
      const unsigned short* pb1 = &lB[cur][(wn * 32 + 1 * 16 + lm) * 40 + hi * 8];
      b1.q[0] = *(const u32x4*)pb1;
      b1.q[1] = *(const u32x4*)(pb1 + 16);
    }
    acc[0][0] = __builtin_amdgcn_wmma_f32_16x16x32_bf16(
        false, a0.v, false, b0.v, (short)0, acc[0][0], false, false);
    acc[0][1] = __builtin_amdgcn_wmma_f32_16x16x32_bf16(
        false, a0.v, false, b1.v, (short)0, acc[0][1], false, false);
    acc[1][0] = __builtin_amdgcn_wmma_f32_16x16x32_bf16(
        false, a1.v, false, b0.v, (short)0, acc[1][0], false, false);
    acc[1][1] = __builtin_amdgcn_wmma_f32_16x16x32_bf16(
        false, a1.v, false, b1.v, (short)0, acc[1][1], false, false);
  }

  // ---- epilogue / store. C layout: lane = col, VGPR r -> row r + 8*hi
#pragma unroll
  for (int i = 0; i < 2; i++) {
#pragma unroll
    for (int j = 0; j < 2; j++) {
      const int col = n_base + wn * 32 + j * 16 + lm;
#pragma unroll
      for (int r = 0; r < 8; r++) {
        const int row = m_base + wm * 32 + i * 16 + hi * 8 + r;
        const size_t idx = (size_t)row * N + col;
        float v = acc[i][j][r];
        if (epi == 1) v += bias[col];
        if (epi == 3) {
          hio[idx] = hio[idx] + v + add2[idx];
        } else {
          C[idx] = v;
        }
      }
    }
  }
}

// ---------------------------------------------------------------------------
// Wave-per-row LayerNorm helpers (D = 512, 32 lanes x 16 elems)
// ---------------------------------------------------------------------------
__device__ __forceinline__ void wave_ln(float* v, int lane,
                                        const float* __restrict__ w,
                                        const float* __restrict__ b) {
  float s = 0.0f;
#pragma unroll
  for (int j = 0; j < 16; j++) s += v[j];
#pragma unroll
  for (int o = 16; o > 0; o >>= 1) s += __shfl_xor(s, o, 32);
  const float mean = s * (1.0f / 512.0f);
  float vs = 0.0f;
#pragma unroll
  for (int j = 0; j < 16; j++) {
    float d = v[j] - mean;
    vs += d * d;
  }
#pragma unroll
  for (int o = 16; o > 0; o >>= 1) vs += __shfl_xor(vs, o, 32);
  const float rstd = rsqrtf(vs * (1.0f / 512.0f) + 1e-5f);
#pragma unroll
  for (int j = 0; j < 16; j++) {
    const int col = lane + 32 * j;
    v[j] = (v[j] - mean) * rstd * w[col] + b[col];
  }
}

// fx = LN(in); writes f32 (for later elementwise use) AND bf16 (GEMM operand)
__global__ __launch_bounds__(256) void ln_kernel(
    const float* __restrict__ in, float* __restrict__ out,
    unsigned short* __restrict__ outb, const float* __restrict__ w,
    const float* __restrict__ b) {
  const int lane = threadIdx.x & 31, wid = threadIdx.x >> 5;
  const size_t row = (size_t)blockIdx.x * 8 + wid;
  const float* x = in + row * DMODEL;
  float v[16];
#pragma unroll
  for (int j = 0; j < 16; j++) v[j] = x[lane + 32 * j];
  wave_ln(v, lane, w, b);
  float* o = out + row * DMODEL;
  unsigned short* ob = outb + row * DMODEL;
#pragma unroll
  for (int j = 0; j < 16; j++) {
    o[lane + 32 * j] = v[j];
    ob[lane + 32 * j] = f2bf(v[j]);
  }
}

// fused: out = LN(LN(h, w1, b1), w2, b2)   (prenorm -> S5Block attn-norm)
__global__ __launch_bounds__(256) void prenorm2_kernel(
    const float* __restrict__ in, float* __restrict__ out,
    unsigned short* __restrict__ outb, const float* __restrict__ w1,
    const float* __restrict__ b1, const float* __restrict__ w2,
    const float* __restrict__ b2) {
  const int lane = threadIdx.x & 31, wid = threadIdx.x >> 5;
  const size_t row = (size_t)blockIdx.x * 8 + wid;
  const float* x = in + row * DMODEL;
  float v[16];
#pragma unroll
  for (int j = 0; j < 16; j++) v[j] = x[lane + 32 * j];
  wave_ln(v, lane, w1, b1);
  wave_ln(v, lane, w2, b2);
  float* o = out + row * DMODEL;
  unsigned short* ob = outb + row * DMODEL;
#pragma unroll
  for (int j = 0; j < 16; j++) {
    o[lane + 32 * j] = v[j];
    ob[lane + 32 * j] = f2bf(v[j]);
  }
}

// ---------------------------------------------------------------------------
// Weight / input prep kernels (f32 -> bf16, layout (N, K) row-major)
// ---------------------------------------------------------------------------
__global__ void cvt_bf16_kernel(const float* __restrict__ src,
                                unsigned short* __restrict__ dst, int n) {
  const int i = (int)blockIdx.x * 256 + (int)threadIdx.x;
  if (i < n) dst[i] = f2bf(src[i]);
}

// transpose enc_w (DIN, D) -> (D, DIN) bf16
__global__ void prep_enc_kernel(const float* __restrict__ ew,
                                unsigned short* __restrict__ dst) {
  const int i = (int)blockIdx.x * 256 + (int)threadIdx.x; // over D*DIN
  const int n = i / DIN, k = i % DIN;
  dst[i] = f2bf(ew[(size_t)k * DMODEL + n]);
}

// per layer: lam_bar = exp(lam*step), coef = (lam_bar - 1)/lam
__global__ void prep_lam_kernel(const float* __restrict__ lr,
                                const float* __restrict__ li,
                                const float* __restrict__ ls,
                                float* __restrict__ LamR,
                                float* __restrict__ LamI,
                                float* __restrict__ Coef) {
  const int p = (int)blockIdx.x * 256 + (int)threadIdx.x;
  if (p >= PDIM) return;
  const float re = lr[p], im = li[p];
  const float step = expf(ls[p]);
  const float er = expf(re * step);
  const float th = im * step;
  const float br = er * cosf(th);
  const float bi = er * sinf(th);
  LamR[p] = br;
  LamI[p] = bi;
  const float inv = 1.0f / (re * re + im * im);
  Coef[p] = ((br - 1.0f) * re + bi * im) * inv;        // Re coef
  Coef[PDIM + p] = (bi * re - (br - 1.0f) * im) * inv; // Im coef
}

// Wb (2P, D) bf16: row p = Re(B_bar[p,:]), row P+p = Im(B_bar[p,:])
__global__ void prep_wb_kernel(const float* __restrict__ Bre,
                               const float* __restrict__ Bim,
                               const float* __restrict__ Coef,
                               unsigned short* __restrict__ Wb) {
  const int i = (int)blockIdx.x * 256 + (int)threadIdx.x; // over P*D
  const int p = i >> 9;
  const float cr = Coef[p], ci = Coef[PDIM + p];
  const float br = Bre[i], bi = Bim[i];
  Wb[i] = f2bf(cr * br - ci * bi);
  Wb[i + PDIM * DMODEL] = f2bf(cr * bi + ci * br);
}

// Wc (D, 2P) bf16: [Cre | -Cim]
__global__ void prep_wc_kernel(const float* __restrict__ Cre,
                               const float* __restrict__ Cim,
                               unsigned short* __restrict__ Wc) {
  const int i = (int)blockIdx.x * 256 + (int)threadIdx.x; // over D*P
  const int d = i >> 9, p = i & 511;
  Wc[(size_t)d * 1024 + p] = f2bf(Cre[i]);
  Wc[(size_t)d * 1024 + 512 + p] = f2bf(-Cim[i]);
}

// ---------------------------------------------------------------------------
// Chunked complex scan: x_t = lam_bar * x_{t-1} + Bu_t
// Buffers layout (M, 1024): re at [p], im at [512+p]
// ---------------------------------------------------------------------------
__global__ __launch_bounds__(256) void scanA_kernel(
    const float* __restrict__ U, float* __restrict__ X,
    float* __restrict__ Carry, const float* __restrict__ LamR,
    const float* __restrict__ LamI) {
  const int b = (int)blockIdx.x / NCH;
  const int c = (int)blockIdx.x % NCH;
#pragma unroll
  for (int s = 0; s < 2; s++) {
    const int p = (int)threadIdx.x + s * 256;
    const float ar = LamR[p], ai = LamI[p];
    float yr = 0.0f, yi = 0.0f;
    size_t base = ((size_t)(b * LSEQ + c * CHUNK)) * 1024 + p;
    for (int t = 0; t < CHUNK; t++) {
      const float ur = U[base], ui = U[base + 512];
      const float nyr = ar * yr - ai * yi + ur;
      const float nyi = ar * yi + ai * yr + ui;
      yr = nyr;
      yi = nyi;
      X[base] = yr;
      X[base + 512] = yi;
      base += 1024;
    }
    const size_t ci = ((size_t)(b * NCH + c)) * 1024 + p;
    Carry[ci] = yr;
    Carry[ci + 512] = yi;
  }
}

__global__ __launch_bounds__(256) void scanB_kernel(
    const float* __restrict__ Carry, float* __restrict__ Prefix,
    const float* __restrict__ LamR, const float* __restrict__ LamI) {
  const int b = (int)blockIdx.x;
#pragma unroll
  for (int s = 0; s < 2; s++) {
    const int p = (int)threadIdx.x + s * 256;
    float pr = LamR[p], pi = LamI[p];
#pragma unroll
    for (int k = 0; k < 6; k++) { // lam_bar^64
      const float nr = pr * pr - pi * pi;
      const float ni = 2.0f * pr * pi;
      pr = nr;
      pi = ni;
    }
    float sr = 0.0f, si = 0.0f;
    for (int c = 0; c < NCH; c++) {
      const size_t idx = ((size_t)(b * NCH + c)) * 1024 + p;
      Prefix[idx] = sr; // state entering chunk c
      Prefix[idx + 512] = si;
      const float cr = Carry[idx], cim = Carry[idx + 512];
      const float nsr = pr * sr - pi * si + cr;
      const float nsi = pr * si + pi * sr + cim;
      sr = nsr;
      si = nsi;
    }
  }
}

// final pass: xs = lam^(t+1)*prefix + y, emitted directly as bf16 GEMM operand
__global__ __launch_bounds__(256) void scanC_kernel(
    const float* __restrict__ X, unsigned short* __restrict__ Xb,
    const float* __restrict__ Prefix, const float* __restrict__ LamR,
    const float* __restrict__ LamI) {
  const int b = (int)blockIdx.x / NCH;
  const int c = (int)blockIdx.x % NCH;
#pragma unroll
  for (int s = 0; s < 2; s++) {
    const int p = (int)threadIdx.x + s * 256;
    const float ar = LamR[p], ai = LamI[p];
    const size_t pidx = ((size_t)(b * NCH + c)) * 1024 + p;
    const float sr = Prefix[pidx], si = Prefix[pidx + 512];
    float apr = ar, api = ai; // lam_bar^(t+1)
    size_t base = ((size_t)(b * LSEQ + c * CHUNK)) * 1024 + p;
    for (int t = 0; t < CHUNK; t++) {
      const float yr = X[base], yi = X[base + 512];
      Xb[base] = f2bf(apr * sr - api * si + yr);
      Xb[base + 512] = f2bf(apr * si + api * sr + yi);
      const float nar = apr * ar - api * ai;
      const float nai = apr * ai + api * ar;
      apr = nar;
      api = nai;
      base += 1024;
    }
  }
}

// ---------------------------------------------------------------------------
// Elementwise fusions
// ---------------------------------------------------------------------------
// z2 = gelu(2*s + Dp[n]*fx1) + fx1   (in place into s)
__global__ __launch_bounds__(256) void s5post_kernel(
    float* __restrict__ s, const float* __restrict__ fx,
    const float* __restrict__ Dp) {
  const size_t i = (size_t)blockIdx.x * 256 + threadIdx.x;
  const int col = (int)(i & 511);
  const float f = fx[i];
  const float val = 2.0f * s[i] + Dp[col] * f;
  s[i] = gelu_exact(val) + f;
}

// q = a * gelu(g) from e (M, 2D); emitted as bf16 (GEMM operand only)
__global__ __launch_bounds__(256) void geglu_kernel(
    const float* __restrict__ e, unsigned short* __restrict__ q) {
  const size_t i = (size_t)blockIdx.x * 256 + threadIdx.x;
  const size_t row = i >> 9;
  const int col = (int)(i & 511);
  const float* er = e + (row << 10);
  q[i] = f2bf(er[col] * gelu_exact(er[512 + col]));
}

// masked mean pool + linear head: out[b] = (sum_{t<len} h[b,t,:].w)/len + hb
__global__ __launch_bounds__(256) void pool_head_kernel(
    const float* __restrict__ H, const int* __restrict__ lens,
    const float* __restrict__ hw, const float* __restrict__ hb,
    float* __restrict__ out) {
  __shared__ float red[256];
  const int b = (int)blockIdx.x;
  int len = lens[b];
  len = len < 1 ? 1 : (len > LSEQ ? LSEQ : len);
  const size_t n = (size_t)len * DMODEL;
  const float* base = H + (size_t)b * LSEQ * DMODEL;
  float acc = 0.0f;
  for (size_t i = threadIdx.x; i < n; i += 256) {
    acc += base[i] * hw[(int)(i & 511)];
  }
  red[threadIdx.x] = acc;
  __syncthreads();
  for (int o = 128; o > 0; o >>= 1) {
    if ((int)threadIdx.x < o) red[threadIdx.x] += red[threadIdx.x + o];
    __syncthreads();
  }
  if (threadIdx.x == 0) out[b] = red[0] / (float)len + hb[0];
}

// ---------------------------------------------------------------------------
// Host-side orchestration
// ---------------------------------------------------------------------------
extern "C" void kernel_launch(void* const* d_in, const int* in_sizes, int n_in,
                              void* d_out, int out_size, void* d_ws,
                              size_t ws_size, hipStream_t stream) {
  (void)in_sizes; (void)n_in; (void)out_size; (void)ws_size;
  const float* x      = (const float*)d_in[0];
  const int*   lens   = (const int*)d_in[1];
  const float* enc_w  = (const float*)d_in[2];
  const float* enc_b  = (const float*)d_in[3];
  const float* lam_re = (const float*)d_in[4];
  const float* lam_im = (const float*)d_in[5];
  const float* Bre    = (const float*)d_in[6];
  const float* Bim    = (const float*)d_in[7];
  const float* Cre    = (const float*)d_in[8];
  const float* Cim    = (const float*)d_in[9];
  const float* Dp     = (const float*)d_in[10];
  const float* lstep  = (const float*)d_in[11];
  const float* an_w   = (const float*)d_in[12];
  const float* an_b   = (const float*)d_in[13];
  const float* fn_w   = (const float*)d_in[14];
  const float* fn_b   = (const float*)d_in[15];
  const float* ffe_w  = (const float*)d_in[16];
  const float* ffd_w  = (const float*)d_in[17];
  const float* norm_w = (const float*)d_in[18];
  const float* norm_b = (const float*)d_in[19];
  const float* head_w = (const float*)d_in[20];
  const float* head_b = (const float*)d_in[21];
  float* out = (float*)d_out;

  const size_t SZ_ACT  = (size_t)MROWS * DMODEL * sizeof(float); // 64 MB
  const size_t SZ_ACT2 = (size_t)MROWS * 1024 * sizeof(float);   // 128 MB
  char* p = (char*)d_ws;
  float* H   = (float*)p; p += SZ_ACT;
  float* FX1 = (float*)p; p += SZ_ACT;
  float* U   = (float*)p; p += SZ_ACT2; // Bu (re|im); reused as E (GEGLU enc)
  float* X   = (float*)p; p += SZ_ACT2; // scan partials
  float* S   = (float*)p; p += SZ_ACT;  // S5 out -> z2 in place
  float* F   = (float*)p; p += SZ_ACT;  // fx2
  // bf16 GEMM A-operands
  unsigned short* XINb = (unsigned short*)p; p += (size_t)MROWS * DIN * 2;
  unsigned short* FX1b = (unsigned short*)p; p += (size_t)MROWS * DMODEL * 2;
  unsigned short* Xb   = (unsigned short*)p; p += (size_t)MROWS * 1024 * 2;
  unsigned short* Fb   = (unsigned short*)p; p += (size_t)MROWS * DMODEL * 2;
  unsigned short* Qb   = (unsigned short*)p; p += (size_t)MROWS * DMODEL * 2;
  // bf16 weights
  unsigned short* Wenc = (unsigned short*)p; p += (size_t)DMODEL * DIN * 2;
  unsigned short* Wb   = (unsigned short*)p; p += (size_t)1024 * DMODEL * 2;
  unsigned short* Wc   = (unsigned short*)p; p += (size_t)DMODEL * 1024 * 2;
  unsigned short* Wffe = (unsigned short*)p; p += (size_t)1024 * DMODEL * 2;
  unsigned short* Wffd = (unsigned short*)p; p += (size_t)DMODEL * DMODEL * 2;
  float* LamR = (float*)p; p += PDIM * 4;
  float* LamI = (float*)p; p += PDIM * 4;
  float* Coef = (float*)p; p += 2 * PDIM * 4;
  float* Carry  = (float*)p; p += (size_t)BDIM * NCH * 1024 * 4;
  float* Prefix = (float*)p; p += (size_t)BDIM * NCH * 1024 * 4;

  const int GB_512  = (MROWS / 128) * (512 / 64);  // GEMM blocks, N=512
  const int GB_1024 = (MROWS / 128) * (1024 / 64); // GEMM blocks, N=1024
  const int EW = (int)(((size_t)MROWS * DMODEL) / 256); // 65536 blocks

  // --- encoder: H = x @ enc_w + enc_b
  cvt_bf16_kernel<<<(MROWS * DIN) / 256, 256, 0, stream>>>(x, XINb,
                                                           MROWS * DIN);
  prep_enc_kernel<<<(DMODEL * DIN) / 256, 256, 0, stream>>>(enc_w, Wenc);
  gemm_bf16_wmma<<<GB_512, 256, 0, stream>>>(XINb, Wenc, H, MROWS, DMODEL, DIN,
                                             1, enc_b, nullptr, nullptr);

  for (int l = 0; l < NLAYER; l++) {
    const size_t oP = (size_t)l * PDIM;
    const size_t oD = (size_t)l * DMODEL;
    const size_t oPD = (size_t)l * PDIM * DMODEL;
    // weight prep for this layer
    prep_lam_kernel<<<2, 256, 0, stream>>>(lam_re + oP, lam_im + oP,
                                           lstep + oP, LamR, LamI, Coef);
    prep_wb_kernel<<<(PDIM * DMODEL) / 256, 256, 0, stream>>>(
        Bre + oPD, Bim + oPD, Coef, Wb);
    prep_wc_kernel<<<(DMODEL * PDIM) / 256, 256, 0, stream>>>(
        Cre + (size_t)l * DMODEL * PDIM, Cim + (size_t)l * DMODEL * PDIM, Wc);
    cvt_bf16_kernel<<<(1024 * DMODEL) / 256, 256, 0, stream>>>(
        ffe_w + (size_t)l * 1024 * DMODEL, Wffe, 1024 * DMODEL);
    cvt_bf16_kernel<<<(DMODEL * DMODEL) / 256, 256, 0, stream>>>(
        ffd_w + (size_t)l * DMODEL * DMODEL, Wffd, DMODEL * DMODEL);

    // fx1 = LN(LN(h, norm), an)
    prenorm2_kernel<<<MROWS / 8, 256, 0, stream>>>(H, FX1, FX1b, norm_w + oD,
                                                   norm_b + oD, an_w + oD,
                                                   an_b + oD);
    // Bu (re|im) = fx1 @ Wb^T   (M x 1024)
    gemm_bf16_wmma<<<GB_1024, 256, 0, stream>>>(FX1b, Wb, U, MROWS, 1024,
                                                DMODEL, 0, nullptr, nullptr,
                                                nullptr);
    // chunked associative scan
    scanA_kernel<<<BDIM * NCH, 256, 0, stream>>>(U, X, Carry, LamR, LamI);
    scanB_kernel<<<BDIM, 256, 0, stream>>>(Carry, Prefix, LamR, LamI);
    scanC_kernel<<<BDIM * NCH, 256, 0, stream>>>(X, Xb, Prefix, LamR, LamI);
    // s = xs_re@Cre^T + xs_im@(-Cim)^T  (doubling and Dp fused next)
    gemm_bf16_wmma<<<GB_512, 256, 0, stream>>>(Xb, Wc, S, MROWS, DMODEL, 1024,
                                               0, nullptr, nullptr, nullptr);
    // z2 = gelu(2*s + Dp*fx1) + fx1  (in place in S)
    s5post_kernel<<<EW, 256, 0, stream>>>(S, FX1, Dp + oD);
    // fx2 = LN(z2)
    ln_kernel<<<MROWS / 8, 256, 0, stream>>>(S, F, Fb, fn_w + oD, fn_b + oD);
    // e = fx2 @ ffe^T  (M x 1024) -> U
    gemm_bf16_wmma<<<GB_1024, 256, 0, stream>>>(Fb, Wffe, U, MROWS, 1024,
                                                DMODEL, 0, nullptr, nullptr,
                                                nullptr);
    // q = a * gelu(g)
    geglu_kernel<<<EW, 256, 0, stream>>>(U, Qb);
    // h += q @ ffd^T + fx2
    gemm_bf16_wmma<<<GB_512, 256, 0, stream>>>(Qb, Wffd, nullptr, MROWS,
                                               DMODEL, DMODEL, 3, nullptr, F,
                                               H);
  }

  pool_head_kernel<<<BDIM, 256, 0, stream>>>(H, lens, head_w, head_b, out);
}
